// HybridChatModel_53755810677316
// MI455X (gfx1250) — matrix operand order repaired and checked by reading, not compiled, not run on hardware
//
#include <hip/hip_runtime.h>
#include <hip/hip_bf16.h>
#include <math.h>

// Model dims
#define BB   32
#define SS   96
#define TT   96
#define DD   256
#define NHH  8
#define DHH  32
#define FFF  768
#define HDD  320
#define VV   32000

typedef _Float16 half8 __attribute__((ext_vector_type(8)));
typedef _Float16 v16h  __attribute__((ext_vector_type(16)));
typedef float    v8f   __attribute__((ext_vector_type(8)));

__device__ __forceinline__ float geluf(float x)    { return 0.5f * x * (1.0f + erff(x * 0.70710678118654752f)); }
__device__ __forceinline__ float sigmoidf(float x) { return 1.0f / (1.0f + expf(-x)); }
__device__ __forceinline__ float wave_max(float v) { for (int o = 16; o > 0; o >>= 1) v = fmaxf(v, __shfl_xor(v, o, 32)); return v; }
__device__ __forceinline__ float wave_sum(float v) { for (int o = 16; o > 0; o >>= 1) v += __shfl_xor(v, o, 32); return v; }

// ---------------------------------------------------------------------------
// fp32 -> fp16 conversion
// ---------------------------------------------------------------------------
__global__ void cvt_f16_kernel(const float* __restrict__ in, _Float16* __restrict__ out, long n) {
    long i = (long)blockIdx.x * blockDim.x + threadIdx.x;
    if (i < n) out[i] = (_Float16)in[i];
}

// ---------------------------------------------------------------------------
// WMMA GEMM (single wave / 16x64 tile). Used for small M (M%16==0, N%64==0).
// ---------------------------------------------------------------------------
__global__ __launch_bounds__(32) void wmma_gemm_kernel(
    const _Float16* __restrict__ A, const _Float16* __restrict__ W,
    const float* __restrict__ bias, float* __restrict__ C,
    int M, int N, int K, int act)
{
    const int  lane  = threadIdx.x & 31;
    const int  l16   = lane & 15;
    const int  hi    = lane >> 4;
    const long mBase = (long)blockIdx.x * 16;
    const long nBase = (long)blockIdx.y * 64;

    const _Float16* arow = A + (mBase + l16) * (long)K;

    v8f acc[4] = { v8f{}, v8f{}, v8f{}, v8f{} };

    for (int k0 = 0; k0 < K; k0 += 32) {
        const _Float16* pa = arow + k0 + 8 * hi;
        half8 a0 = *(const half8*)(pa);
        half8 a1 = *(const half8*)(pa + 16);
        v16h af;
#pragma unroll
        for (int i = 0; i < 8; ++i) { af[i] = a0[i]; af[i + 8] = a1[i]; }

#pragma unroll
        for (int j = 0; j < 4; ++j) {
            const _Float16* pb = W + (nBase + j * 16 + l16) * (long)K + k0 + 8 * hi;
            half8 b0 = *(const half8*)(pb);
            half8 b1 = *(const half8*)(pb + 16);
            v16h bf;
#pragma unroll
            for (int i = 0; i < 8; ++i) { bf[i] = b0[i]; bf[i + 8] = b1[i]; }

            acc[j] = __builtin_amdgcn_wmma_f32_16x16x32_f16(
                false, af, false, bf, (short)0, acc[j], false, false);
        }
    }

#pragma unroll
    for (int j = 0; j < 4; ++j) {
        long  ncol = nBase + j * 16 + l16;
        float bv   = bias ? bias[ncol] : 0.0f;
#pragma unroll
        for (int r = 0; r < 8; ++r) {
            long  row = mBase + r + 8 * hi;
            float v   = acc[j][r] + bv;
            if (act == 1) v = geluf(v);
            C[row * (long)N + ncol] = v;
        }
    }
}

// ---------------------------------------------------------------------------
// WMMA GEMM with async-to-LDS B staging (128x64 block tile, 4 waves).
// Wave w computes rows [mBase+32w, +32); B tile (64 x 32 f16) is staged in
// LDS via global_load_async_to_lds_b128 (ASYNCcnt), double-buffered.
// The LDS destination offset is derived from &smem[...] (ptrtoint) so the
// array provably escapes into the asm: fragment ds_loads cannot be folded,
// and the low 32 bits of the generic pointer ARE the LDS offset (ISA 10.2).
// Requires: M%128==0, N%64==0, K%32==0.
// ---------------------------------------------------------------------------
#define LPAD8 5                          // row stride in half8 units (40 halves, 80B)

__global__ __launch_bounds__(128) void wmma_gemm_lds_kernel(
    const _Float16* __restrict__ A, const _Float16* __restrict__ W,
    const float* __restrict__ bias, float* __restrict__ C,
    int M, int N, int K, int act)
{
    __shared__ half8 smem[2][64 * LPAD8];

    const int  tid   = threadIdx.x;
    const int  lane  = tid & 31;
    const int  wave  = tid >> 5;            // 0..3
    const int  l16   = lane & 15;
    const int  hi    = lane >> 4;
    const long mBase = (long)blockIdx.x * 128;
    const long nBase = (long)blockIdx.y * 64;
    const int  nk    = K / 32;

    // stage one 64x32 f16 B tile into smem[buf] (each thread: 2 async b128 copies)
    auto stage = [&](int kstep, int buf) {
        const int k0 = kstep * 32;
#pragma unroll
        for (int i = 0; i < 2; ++i) {
            int c  = tid + i * 128;          // chunk 0..255
            int nr = c >> 2;                 // B row 0..63
            int kc = c & 3;                  // half8 column 0..3
            unsigned long long gaddr =
                (unsigned long long)(const void*)(W + (nBase + nr) * (long)K + k0 + kc * 8);
            // LDS byte offset from the shared object's own address (captures smem)
            unsigned loff = (unsigned)(unsigned long long)&smem[buf][nr * LPAD8 + kc];
            asm volatile("global_load_async_to_lds_b128 %0, %1, off"
                         :: "v"(loff), "v"(gaddr) : "memory");
        }
    };

    const _Float16* arow0 = A + (mBase + wave * 32 + l16) * (long)K;
    const _Float16* arow1 = arow0 + 16 * (long)K;

    v8f acc0[4] = { v8f{}, v8f{}, v8f{}, v8f{} };
    v8f acc1[4] = { v8f{}, v8f{}, v8f{}, v8f{} };

    stage(0, 0);
    for (int i = 0; i < nk; ++i) {
        const int buf = i & 1;
        asm volatile("s_wait_asynccnt 0x0" ::: "memory");  // my staging chunks landed
        __syncthreads();                                   // everyone's chunks landed
        if (i + 1 < nk) stage(i + 1, buf ^ 1);             // prefetch next tile (async)

        const int k0 = i * 32;
        const _Float16* pa0 = arow0 + k0 + 8 * hi;
        const _Float16* pa1 = arow1 + k0 + 8 * hi;
        half8 a0l = *(const half8*)(pa0);
        half8 a0h = *(const half8*)(pa0 + 16);
        half8 a1l = *(const half8*)(pa1);
        half8 a1h = *(const half8*)(pa1 + 16);
        v16h af0, af1;
#pragma unroll
        for (int e = 0; e < 8; ++e) {
            af0[e] = a0l[e]; af0[e + 8] = a0h[e];
            af1[e] = a1l[e]; af1[e + 8] = a1h[e];
        }

#pragma unroll
        for (int j = 0; j < 4; ++j) {
            // direct shared-array indexing -> ds_load_b128
            int rbase = (j * 16 + l16) * LPAD8 + hi;
            half8 b0 = smem[buf][rbase];
            half8 b1 = smem[buf][rbase + 2];
            v16h bf;
#pragma unroll
            for (int e = 0; e < 8; ++e) { bf[e] = b0[e]; bf[e + 8] = b1[e]; }
            acc0[j] = __builtin_amdgcn_wmma_f32_16x16x32_f16(false, af0, false, bf, (short)0, acc0[j], false, false);
            acc1[j] = __builtin_amdgcn_wmma_f32_16x16x32_f16(false, af1, false, bf, (short)0, acc1[j], false, false);
        }
    }

#pragma unroll
    for (int j = 0; j < 4; ++j) {
        long  ncol = nBase + j * 16 + l16;
        float bv   = bias ? bias[ncol] : 0.0f;
#pragma unroll
        for (int r = 0; r < 8; ++r) {
            long row0 = mBase + wave * 32 + r + 8 * hi;
            float v0 = acc0[j][r] + bv;
            if (act == 1) v0 = geluf(v0);
            C[row0 * (long)N + ncol] = v0;
            long row1 = row0 + 16;
            float v1 = acc1[j][r] + bv;
            if (act == 1) v1 = geluf(v1);
            C[row1 * (long)N + ncol] = v1;
        }
    }
}

// ---------------------------------------------------------------------------
// Embedding: h[b,s,:] = tokEmb[x[b,s]] * sqrt(D) + posEmb[s]
// ---------------------------------------------------------------------------
__global__ void embed_kernel(const int* __restrict__ x, const float* __restrict__ tokEmb,
                             const float* __restrict__ posEmb, float* __restrict__ h) {
    long idx = (long)blockIdx.x * blockDim.x + threadIdx.x;
    if (idx >= (long)BB * SS * DD) return;
    int d  = idx % DD;
    long bs = idx / DD;
    int s  = bs % SS;
    int tok = x[bs];
    h[idx] = tokEmb[(long)tok * DD + d] * 16.0f + posEmb[(long)s * DD + d];
}

// ---------------------------------------------------------------------------
// Encoder attention scores: sc[b,h,i,j] = <q,k>/sqrt(DH), masked
// ---------------------------------------------------------------------------
__global__ void enc_score_kernel(const float* __restrict__ qkv, const int* __restrict__ x,
                                 float* __restrict__ sc) {
    long idx = (long)blockIdx.x * blockDim.x + threadIdx.x;
    if (idx >= (long)BB * NHH * SS * SS) return;
    int j = idx % SS; long t = idx / SS;
    int i = t % SS;  t /= SS;
    int hh = t % NHH;
    int b  = t / NHH;
    const float* q = qkv + ((long)(b * SS + i)) * (3 * DD) + hh * DHH;
    const float* k = qkv + ((long)(b * SS + j)) * (3 * DD) + DD + hh * DHH;
    float s = 0.0f;
#pragma unroll
    for (int d = 0; d < DHH; ++d) s += q[d] * k[d];
    s *= 0.17677669529663689f;                       // 1/sqrt(32)
    if (x[b * SS + j] == 0) s = -1.0e9f;
    sc[idx] = s;
}

// ---------------------------------------------------------------------------
// Row softmax (one wave per row)
// ---------------------------------------------------------------------------
__global__ __launch_bounds__(32) void softmax_rows_kernel(float* __restrict__ p, int L) {
    long row = blockIdx.x;
    float* r = p + row * (long)L;
    int lane = threadIdx.x;
    float mx = -1.0e30f;
    for (int j = lane; j < L; j += 32) mx = fmaxf(mx, r[j]);
    mx = wave_max(mx);
    float sum = 0.0f;
    for (int j = lane; j < L; j += 32) { float e = expf(r[j] - mx); r[j] = e; sum += e; }
    sum = wave_sum(sum);
    float inv = 1.0f / sum;
    for (int j = lane; j < L; j += 32) r[j] *= inv;
}

// ---------------------------------------------------------------------------
// Encoder attention output: a[b,i,h*DH+d] = sum_j p[b,h,i,j] * v[b,j,h,d]
// ---------------------------------------------------------------------------
__global__ void enc_attnout_kernel(const float* __restrict__ sc, const float* __restrict__ qkv,
                                   float* __restrict__ out) {
    long idx = (long)blockIdx.x * blockDim.x + threadIdx.x;
    if (idx >= (long)BB * SS * DD) return;
    int dd = idx % DD;
    int h  = dd / DHH;
    int d  = dd % DHH;
    long bs = idx / DD;
    int i = bs % SS;
    int b = bs / SS;
    const float* prow = sc + ((long)((b * NHH + h) * SS + i)) * SS;
    float s = 0.0f;
    for (int j = 0; j < SS; ++j)
        s += prow[j] * qkv[(long)(b * SS + j) * (3 * DD) + 2 * DD + h * DHH + d];
    out[idx] = s;
}

// ---------------------------------------------------------------------------
// LayerNorm over last dim (one wave per row). out = LN(x [+ res]) * g + b
// ---------------------------------------------------------------------------
__global__ __launch_bounds__(32) void ln_kernel(const float* __restrict__ x, const float* __restrict__ res,
                                                const float* __restrict__ g, const float* __restrict__ b,
                                                float* __restrict__ out, int D) {
    long row = blockIdx.x;
    int lane = threadIdx.x;
    const float* xr = x + row * (long)D;
    const float* rr = res ? res + row * (long)D : nullptr;
    float s = 0.0f;
    for (int d = lane; d < D; d += 32) s += xr[d] + (rr ? rr[d] : 0.0f);
    float mean = wave_sum(s) / (float)D;
    float v = 0.0f;
    for (int d = lane; d < D; d += 32) {
        float t = xr[d] + (rr ? rr[d] : 0.0f) - mean;
        v += t * t;
    }
    float var = wave_sum(v) / (float)D;
    float inv = rsqrtf(var + 1e-5f);
    for (int d = lane; d < D; d += 32) {
        float t = xr[d] + (rr ? rr[d] : 0.0f);
        out[row * (long)D + d] = (t - mean) * inv * g[d] + b[d];
    }
}

// ---------------------------------------------------------------------------
// Masked mean pool: pooled[b,d] = sum_s mem[b,s,d]*(x!=0) / max(len,1)
// ---------------------------------------------------------------------------
__global__ void pool_kernel(const float* __restrict__ mem, const int* __restrict__ x,
                            const int* __restrict__ lengths, float* __restrict__ pooled) {
    long idx = (long)blockIdx.x * blockDim.x + threadIdx.x;
    if (idx >= (long)BB * DD) return;
    int d = idx % DD;
    int b = idx / DD;
    float s = 0.0f;
    for (int t = 0; t < SS; ++t)
        if (x[b * SS + t] != 0) s += mem[((long)(b * SS + t)) * DD + d];
    int len = lengths[b]; if (len < 1) len = 1;
    pooled[idx] = s / (float)len;
}

// ---------------------------------------------------------------------------
// Small GEMM (fp32): C[b,n] = A[b,:]@W[n,:] + bias[n], act 0=none 2=tanh
// ---------------------------------------------------------------------------
__global__ void sgemm_kernel(const float* __restrict__ A, int lda,
                             const float* __restrict__ W, const float* __restrict__ bias,
                             float* __restrict__ C, int ldc, int M, int N, int K, int act) {
    long idx = (long)blockIdx.x * blockDim.x + threadIdx.x;
    if (idx >= (long)M * N) return;
    int n = idx % N;
    int b = idx / N;
    float s = bias ? bias[n] : 0.0f;
    const float* ar = A + (long)b * lda;
    const float* wr = W + (long)n * K;
    for (int k = 0; k < K; ++k) s += ar[k] * wr[k];
    if (act == 2) s = tanhf(s);
    C[(long)b * ldc + n] = s;
}

// ---------------------------------------------------------------------------
// LSTM gates: G[b,n] = X@Wih[n] + Bih[n] + H@Whh[n] + Bhh[n]
// ---------------------------------------------------------------------------
__global__ void lstm_gates_kernel(const float* __restrict__ X, int ldx, int Kx,
                                  const float* __restrict__ Wih, const float* __restrict__ Bih,
                                  const float* __restrict__ H, int ldh, int Kh,
                                  const float* __restrict__ Whh, const float* __restrict__ Bhh,
                                  float* __restrict__ G, int N, int M) {
    long idx = (long)blockIdx.x * blockDim.x + threadIdx.x;
    if (idx >= (long)M * N) return;
    int n = idx % N;
    int b = idx / N;
    float s = Bih[n] + Bhh[n];
    const float* xr = X + (long)b * ldx;
    const float* w1 = Wih + (long)n * Kx;
    for (int k = 0; k < Kx; ++k) s += xr[k] * w1[k];
    const float* hr = H + (long)b * ldh;
    const float* w2 = Whh + (long)n * Kh;
    for (int k = 0; k < Kh; ++k) s += hr[k] * w2[k];
    G[idx] = s;
}

// ---------------------------------------------------------------------------
// LSTM pointwise update (gates order i,f,g,o). H/C updated in place.
// ---------------------------------------------------------------------------
__global__ void lstm_update_kernel(const float* __restrict__ G, float* __restrict__ H, int ldh,
                                   float* __restrict__ Cc, int ldc, int HDim, int M) {
    long idx = (long)blockIdx.x * blockDim.x + threadIdx.x;
    if (idx >= (long)M * HDim) return;
    int j = idx % HDim;
    int b = idx / HDim;
    const float* g = G + (long)b * (4 * HDim);
    float ig = sigmoidf(g[j]);
    float fg = sigmoidf(g[HDim + j]);
    float gg = tanhf(g[2 * HDim + j]);
    float og = sigmoidf(g[3 * HDim + j]);
    float c = fg * Cc[(long)b * ldc + j] + ig * gg;
    Cc[(long)b * ldc + j] = c;
    H[(long)b * ldh + j]  = og * tanhf(c);
}

// ---------------------------------------------------------------------------
// Decoder step helpers
// ---------------------------------------------------------------------------
__global__ void xcat_kernel(const int* __restrict__ resp, const float* __restrict__ decEmb,
                            const float* __restrict__ ctx, float* __restrict__ xcat, int t) {
    long idx = (long)blockIdx.x * blockDim.x + threadIdx.x;
    if (idx >= (long)BB * 2 * DD) return;
    int j = idx % (2 * DD);
    int b = idx / (2 * DD);
    if (j < DD) {
        int tok = resp[b * TT + t];
        xcat[idx] = decEmb[(long)tok * DD + j];
    } else {
        xcat[idx] = ctx[(long)b * DD + (j - DD)];
    }
}

__global__ void dec_score_kernel(const float* __restrict__ q, const float* __restrict__ mem,
                                 const int* __restrict__ x, float* __restrict__ sc) {
    long idx = (long)blockIdx.x * blockDim.x + threadIdx.x;
    if (idx >= (long)BB * SS) return;
    int s = idx % SS;
    int b = idx / SS;
    if (x[b * SS + s] == 0) { sc[idx] = -1.0e9f; return; }
    const float* qr = q + (long)b * DD;
    const float* mr = mem + ((long)(b * SS + s)) * DD;
    float acc = 0.0f;
    for (int d = 0; d < DD; ++d) acc += qr[d] * mr[d];
    sc[idx] = acc;
}

__global__ void dec_ctx_kernel(const float* __restrict__ p, const float* __restrict__ mem,
                               float* __restrict__ ctx) {
    long idx = (long)blockIdx.x * blockDim.x + threadIdx.x;
    if (idx >= (long)BB * DD) return;
    int d = idx % DD;
    int b = idx / DD;
    float s = 0.0f;
    for (int t = 0; t < SS; ++t)
        s += p[b * SS + t] * mem[((long)(b * SS + t)) * DD + d];
    ctx[idx] = s;
}

__global__ void store_h2ctx_kernel(const float* __restrict__ h2, int ldh,
                                   const float* __restrict__ ctx,
                                   float* __restrict__ dst, int t) {
    long idx = (long)blockIdx.x * blockDim.x + threadIdx.x;
    const int CW = HDD + DD; // 576
    if (idx >= (long)BB * CW) return;
    int j = idx % CW;
    int b = idx / CW;
    float v = (j < HDD) ? h2[(long)b * ldh + j] : ctx[(long)b * DD + (j - HDD)];
    dst[((long)(b * TT + t)) * CW + j] = v;
}

// ===========================================================================
extern "C" void kernel_launch(void* const* d_in, const int* in_sizes, int n_in,
                              void* d_out, int out_size, void* d_ws, size_t ws_size,
                              hipStream_t stream) {
    (void)in_sizes; (void)n_in; (void)out_size; (void)ws_size;

    // ----- inputs (setup_inputs dict order, params flattened depth-first) -----
    const int*   x        = (const int*)d_in[0];
    const int*   lengths  = (const int*)d_in[1];
    const int*   resp     = (const int*)d_in[2];
    const float* tokEmb   = (const float*)d_in[3];
    const float* posEmb   = (const float*)d_in[4];
    const float* sh_ln_g  = (const float*)d_in[53];
    const float* sh_ln_b  = (const float*)d_in[54];
    const float* sh_w     = (const float*)d_in[55];
    const float* sh_b     = (const float*)d_in[56];
    const float* intent_w = (const float*)d_in[57];
    const float* intent_b = (const float*)d_in[58];
    const float* style_w  = (const float*)d_in[59];
    const float* style_b  = (const float*)d_in[60];
    const float* cap_w    = (const float*)d_in[61];
    const float* cap_b    = (const float*)d_in[62];
    const float* op_w     = (const float*)d_in[63];
    const float* op_b     = (const float*)d_in[64];
    const float* decEmb   = (const float*)d_in[65];
    const float* l0_wih   = (const float*)d_in[66];
    const float* l0_whh   = (const float*)d_in[67];
    const float* l0_bih   = (const float*)d_in[68];
    const float* l0_bhh   = (const float*)d_in[69];
    const float* l1_wih   = (const float*)d_in[70];
    const float* l1_whh   = (const float*)d_in[71];
    const float* l1_bih   = (const float*)d_in[72];
    const float* l1_bhh   = (const float*)d_in[73];
    const float* initH_w  = (const float*)d_in[74];
    const float* initH_b  = (const float*)d_in[75];
    const float* initC_w  = (const float*)d_in[76];
    const float* initC_b  = (const float*)d_in[77];
    const float* q_w      = (const float*)d_in[78];
    const float* q_b      = (const float*)d_in[79];
    const float* out_w    = (const float*)d_in[80];
    const float* out_b    = (const float*)d_in[81];

    float* outp = (float*)d_out;
    float* out_intent = outp + 0;
    float* out_style  = outp + 1024;
    float* out_cap    = outp + 1280;
    float* out_op     = outp + 1792;
    float* out_shared = outp + 2560;
    float* out_logits = outp + 10752;

    // ----- workspace layout -----
    float* ws = (float*)d_ws;
    float* hbuf   = ws; ws += (long)BB * SS * DD;            // 786432
    float* tmp1   = ws; ws += (long)BB * SS * 3 * DD;        // 2359296 (qkv / ff1)
    float* tmp2   = ws; ws += (long)BB * SS * DD;            // 786432  (attn/proj/ff2)
    float* scores = ws; ws += (long)BB * NHH * SS * SS;      // 2359296
    float* pooled = ws; ws += (long)BB * DD;
    float* lnp    = ws; ws += (long)BB * DD;
    float* h0buf  = ws; ws += (long)BB * 2 * HDD;            // h1 | h2
    float* c0buf  = ws; ws += (long)BB * 2 * HDD;            // c1 | c2
    float* ctx    = ws; ws += (long)BB * DD;
    float* xcat   = ws; ws += (long)BB * 2 * DD;
    float* gbuf   = ws; ws += (long)BB * 4 * HDD;
    float* qbuf   = ws; ws += (long)BB * DD;
    float* dsc    = ws; ws += (long)BB * SS;
    float* h2ctx  = ws; ws += (long)BB * TT * (HDD + DD);    // 1769472
    _Float16* A16 = (_Float16*)ws; ws += ((long)BB * SS * 3 * DD) / 2;  // 2359296 halves
    _Float16* W16 = (_Float16*)ws;                                      // up to 18432000 halves

    auto g1 = [](long n) { return dim3((unsigned)((n + 255) / 256)); };

    // gemm with f16 conversion; async-LDS tile kernel when M allows
    auto gemm = [&](const float* A, const float* W, const float* bias, float* C,
                    int M, int N, int K, int act) {
        long na = (long)M * K, nw = (long)N * K;
        cvt_f16_kernel<<<g1(na), 256, 0, stream>>>(A, A16, na);
        cvt_f16_kernel<<<g1(nw), 256, 0, stream>>>(W, W16, nw);
        if ((M % 128) == 0)
            wmma_gemm_lds_kernel<<<dim3(M / 128, N / 64), 128, 0, stream>>>(A16, W16, bias, C, M, N, K, act);
        else
            wmma_gemm_kernel<<<dim3(M / 16, N / 64), 32, 0, stream>>>(A16, W16, bias, C, M, N, K, act);
    };

    const int  M  = BB * SS;               // 3072 token rows
    const long nh = (long)M * DD;

    // ----- embedding -----
    embed_kernel<<<g1(nh), 256, 0, stream>>>(x, tokEmb, posEmb, hbuf);

    // ----- encoder layers -----
    for (int l = 0; l < 4; ++l) {
        const int base = 5 + 12 * l;
        const float* in_w  = (const float*)d_in[base + 0];
        const float* in_b  = (const float*)d_in[base + 1];
        const float* ow    = (const float*)d_in[base + 2];
        const float* ob    = (const float*)d_in[base + 3];
        const float* ln1_g = (const float*)d_in[base + 4];
        const float* ln1_b = (const float*)d_in[base + 5];
        const float* ln2_g = (const float*)d_in[base + 6];
        const float* ln2_b = (const float*)d_in[base + 7];
        const float* ff_w1 = (const float*)d_in[base + 8];
        const float* ff_b1 = (const float*)d_in[base + 9];
        const float* ff_w2 = (const float*)d_in[base + 10];
        const float* ff_b2 = (const float*)d_in[base + 11];

        gemm(hbuf, in_w, in_b, tmp1, M, 3 * DD, DD, 0);                       // qkv
        enc_score_kernel<<<g1((long)BB * NHH * SS * SS), 256, 0, stream>>>(tmp1, x, scores);
        softmax_rows_kernel<<<dim3(BB * NHH * SS), 32, 0, stream>>>(scores, SS);
        enc_attnout_kernel<<<g1(nh), 256, 0, stream>>>(scores, tmp1, tmp2);   // attn out
        gemm(tmp2, ow, ob, tmp2, M, DD, DD, 0);                                // out proj
        ln_kernel<<<dim3(M), 32, 0, stream>>>(hbuf, tmp2, ln1_g, ln1_b, hbuf, DD);
        gemm(hbuf, ff_w1, ff_b1, tmp1, M, FFF, DD, 1);                         // ff1 + gelu
        gemm(tmp1, ff_w2, ff_b2, tmp2, M, DD, FFF, 0);                         // ff2
        ln_kernel<<<dim3(M), 32, 0, stream>>>(hbuf, tmp2, ln2_g, ln2_b, hbuf, DD);
    }
    // hbuf == memory from here on

    // ----- pooling + shared + heads -----
    pool_kernel<<<g1((long)BB * DD), 256, 0, stream>>>(hbuf, x, lengths, pooled);
    ln_kernel<<<dim3(BB), 32, 0, stream>>>(pooled, nullptr, sh_ln_g, sh_ln_b, lnp, DD);
    gemm(lnp, sh_w, sh_b, out_shared, BB, DD, DD, 1);                          // shared = gelu(...)

    sgemm_kernel<<<g1((long)BB * 32), 256, 0, stream>>>(out_shared, DD, intent_w, intent_b, out_intent, 32, BB, 32, DD, 0);
    sgemm_kernel<<<g1((long)BB * 8),  256, 0, stream>>>(out_shared, DD, style_w,  style_b,  out_style,  8,  BB, 8,  DD, 0);
    sgemm_kernel<<<g1((long)BB * 16), 256, 0, stream>>>(out_shared, DD, cap_w,    cap_b,    out_cap,    16, BB, 16, DD, 0);
    sgemm_kernel<<<g1((long)BB * 24), 256, 0, stream>>>(out_shared, DD, op_w,     op_b,     out_op,     24, BB, 24, DD, 0);

    // ----- decoder init -----
    sgemm_kernel<<<g1((long)BB * 2 * HDD), 256, 0, stream>>>(out_shared, DD, initH_w, initH_b, h0buf, 2 * HDD, BB, 2 * HDD, DD, 2);
    sgemm_kernel<<<g1((long)BB * 2 * HDD), 256, 0, stream>>>(out_shared, DD, initC_w, initC_b, c0buf, 2 * HDD, BB, 2 * HDD, DD, 2);
    hipMemsetAsync(ctx, 0, (size_t)BB * DD * sizeof(float), stream);

    float* h1 = h0buf;        float* c1 = c0buf;        // ld = 640
    float* h2 = h0buf + HDD;  float* c2 = c0buf + HDD;

    // ----- decoder recurrence (logits deferred to one big WMMA GEMM) -----
    for (int t = 0; t < TT; ++t) {
        xcat_kernel<<<g1((long)BB * 2 * DD), 256, 0, stream>>>(resp, decEmb, ctx, xcat, t);
        lstm_gates_kernel<<<g1((long)BB * 4 * HDD), 256, 0, stream>>>(
            xcat, 2 * DD, 2 * DD, l0_wih, l0_bih, h1, 2 * HDD, HDD, l0_whh, l0_bhh, gbuf, 4 * HDD, BB);
        lstm_update_kernel<<<g1((long)BB * HDD), 256, 0, stream>>>(gbuf, h1, 2 * HDD, c1, 2 * HDD, HDD, BB);
        lstm_gates_kernel<<<g1((long)BB * 4 * HDD), 256, 0, stream>>>(
            h1, 2 * HDD, HDD, l1_wih, l1_bih, h2, 2 * HDD, HDD, l1_whh, l1_bhh, gbuf, 4 * HDD, BB);
        lstm_update_kernel<<<g1((long)BB * HDD), 256, 0, stream>>>(gbuf, h2, 2 * HDD, c2, 2 * HDD, HDD, BB);
        sgemm_kernel<<<g1((long)BB * DD), 256, 0, stream>>>(h2, 2 * HDD, q_w, q_b, qbuf, DD, BB, DD, HDD, 0);
        dec_score_kernel<<<g1((long)BB * SS), 256, 0, stream>>>(qbuf, hbuf, x, dsc);
        softmax_rows_kernel<<<dim3(BB), 32, 0, stream>>>(dsc, SS);
        dec_ctx_kernel<<<g1((long)BB * DD), 256, 0, stream>>>(dsc, hbuf, ctx);
        store_h2ctx_kernel<<<g1((long)BB * (HDD + DD)), 256, 0, stream>>>(h2, 2 * HDD, ctx, h2ctx, t);
    }

    // ----- final logits: (B*T, 576) @ out_w^T (V,576) -> (B*T, V), the dominant GEMM -----
    gemm(h2ctx, out_w, out_b, out_logits, BB * TT, VV, HDD + DD, 0);
}
